// TinyGPT_17927193493695
// MI455X (gfx1250) — matrix operand order repaired and checked by reading, not compiled, not run on hardware
//
#include <hip/hip_runtime.h>
#include <stdint.h>

// ---------------------------------------------------------------------------
// TinyGPT forward on gfx1250 (MI455X): bf16 WMMA everywhere, fp32 residual.
// ---------------------------------------------------------------------------
#define V_SZ  8192
#define B_SZ  32
#define T_SZ  256
#define E_SZ  512
#define H_SZ  8
#define D_SZ  64
#define L_SZ  12
#define FF_SZ 2048
#define M_TOK (B_SZ * T_SZ)   // 8192 token rows

typedef __attribute__((ext_vector_type(16))) __bf16 v16bf;
typedef __attribute__((ext_vector_type(8)))  float  v8f;
typedef __attribute__((ext_vector_type(4)))  unsigned int u32x4;
typedef __attribute__((ext_vector_type(4)))  float  f32x4;

union FragAB {          // one WMMA A or B operand (16 bf16 per lane = 32B)
    v16bf  v;
    u32x4  q[2];        // two 16B halves
    __bf16 h[16];
};

__device__ __forceinline__ v8f vzero8() {
    v8f z;
#pragma unroll
    for (int i = 0; i < 8; ++i) z[i] = 0.0f;
    return z;
}

__device__ __forceinline__ __bf16 to_bf16(float f) {
    union { float f; unsigned int u; } a; a.f = f;
    unsigned int u = a.u + 0x7fffu + ((a.u >> 16) & 1u);   // round-to-nearest-even
    union { unsigned short s; __bf16 b; } r;
    r.s = (unsigned short)(u >> 16);
    return r.b;
}

__device__ __forceinline__ v8f wmma_bf16(const v16bf& a, const v16bf& b, v8f c) {
    // D = A(16x32 bf16) x B(32x16 bf16) + C(16x16 f32)
    return __builtin_amdgcn_wmma_f32_16x16x32_bf16(
        /*neg_a=*/false, a, /*neg_b=*/false, b,
        /*c_mod=*/(short)0, c, /*reuse_a=*/false, /*reuse_b=*/false);
}

// ---------------------------------------------------------------------------
// fp32 -> bf16 bulk convert (weights)
// ---------------------------------------------------------------------------
__global__ void k_f32_to_bf16(const float* __restrict__ src,
                              __bf16* __restrict__ dst, long long n) {
    long long i = (long long)blockIdx.x * blockDim.x + threadIdx.x;
    if (i < n) dst[i] = to_bf16(src[i]);
}

// ---------------------------------------------------------------------------
// Embedding: x[b,t,:] = tok_emb[idx[b,t],:] + pos_emb[t,:]   (fp32 residual)
// ---------------------------------------------------------------------------
__global__ void k_embed(const int* __restrict__ idx,
                        const float* __restrict__ tok,
                        const float* __restrict__ pos,
                        float* __restrict__ x) {
    long long i = (long long)blockIdx.x * blockDim.x + threadIdx.x;
    int e  = (int)(i % E_SZ);
    long long bt = i / E_SZ;
    int t  = (int)(bt % T_SZ);
    x[i] = tok[(size_t)idx[bt] * E_SZ + e] + pos[(size_t)t * E_SZ + e];
}

// ---------------------------------------------------------------------------
// LayerNorm over E=512, one wave32 per row, bf16 output.
// ---------------------------------------------------------------------------
__global__ __launch_bounds__(256)
void k_ln_bf16(const float* __restrict__ x, const float* __restrict__ g,
               const float* __restrict__ b, __bf16* __restrict__ out) {
    const int row  = blockIdx.x * 8 + (threadIdx.x >> 5);
    const int lane = threadIdx.x & 31;
    const float* xr = x + (size_t)row * E_SZ;

    float v[16];
    const f32x4* x4 = (const f32x4*)xr;
#pragma unroll
    for (int i = 0; i < 4; ++i) {
        f32x4 f = x4[lane * 4 + i];
#pragma unroll
        for (int j = 0; j < 4; ++j) v[i * 4 + j] = f[j];
    }
    float s = 0.f, s2 = 0.f;
#pragma unroll
    for (int i = 0; i < 16; ++i) { s += v[i]; s2 += v[i] * v[i]; }
#pragma unroll
    for (int m = 16; m >= 1; m >>= 1) {
        s  += __shfl_xor(s,  m, 32);
        s2 += __shfl_xor(s2, m, 32);
    }
    const float mean = s * (1.0f / E_SZ);
    const float var  = s2 * (1.0f / E_SZ) - mean * mean;
    const float rstd = rsqrtf(var + 1e-5f);

    __bf16* orow = out + (size_t)row * E_SZ;
#pragma unroll
    for (int i = 0; i < 16; ++i) {
        const int c = lane * 16 + i;
        orow[c] = to_bf16((v[i] - mean) * rstd * g[c] + b[c]);
    }
}

// ---------------------------------------------------------------------------
// Generic WMMA GEMM: C[M,N] = A[M,K](bf16,rm) x W[K,N](bf16,rm) + bias
// 256 threads = 8 waves; block tile 128x64; each wave a 32x32 tile.
// EPI: 0 = bf16 out, 1 = f32 out + residual, 2 = f32 out, 3 = bf16 gelu out
// ---------------------------------------------------------------------------
template <int EPI>
__global__ __launch_bounds__(256)
void k_gemm(const __bf16* __restrict__ A, const __bf16* __restrict__ W,
            const float* __restrict__ bias, const float* __restrict__ resid,
            __bf16* __restrict__ outb, float* __restrict__ outf,
            int M, int N, int K) {
    __shared__ alignas(16) __bf16 As[128][40];   // [m][k], pad 32->40
    __shared__ alignas(16) __bf16 Bs[64][40];    // [n][k] (transposed), pad

    const int tid  = threadIdx.x;
    const int wave = tid >> 5, lane = tid & 31;
    const int hl   = lane >> 4, lm = lane & 15;
    const int wm   = wave >> 1, wn = wave & 1;       // 4 x 2 wave grid
    const int m0   = blockIdx.y * 128, n0 = blockIdx.x * 64;

    v8f acc[2][2];
#pragma unroll
    for (int mt = 0; mt < 2; ++mt)
#pragma unroll
        for (int nt = 0; nt < 2; ++nt) acc[mt][nt] = vzero8();

    const int arow = tid >> 1;            // 0..127
    const int acol = (tid & 1) * 16;      // 0 / 16
    const int brow = tid >> 3;            // 0..31  (k)
    const int bcol = (tid & 7) * 8;       // 0..56  (n)

    const __bf16* aptr = A + (size_t)(m0 + arow) * K + acol;
    const __bf16* bptr = W + (size_t)brow * N + n0 + bcol;

    for (int kt = 0; kt < K; kt += 32) {
        // stage A tile 128x32
        u32x4 a0 = *(const u32x4*)(aptr + kt);
        u32x4 a1 = *(const u32x4*)(aptr + kt + 8);
        *(u32x4*)&As[arow][acol]     = a0;
        *(u32x4*)&As[arow][acol + 8] = a1;
        // stage B tile 32x64, transposed into [n][k]
        union { u32x4 u; __bf16 h[8]; } wrow;
        wrow.u = *(const u32x4*)(bptr + (size_t)kt * N);
#pragma unroll
        for (int i = 0; i < 8; ++i) Bs[bcol + i][brow] = wrow.h[i];
        __syncthreads();

        if (kt + 32 < K) {  // global_prefetch_b8 the next K-slab
            __builtin_prefetch(aptr + kt + 32, 0, 1);
            __builtin_prefetch(bptr + (size_t)(kt + 32) * N, 0, 1);
        }

        FragAB fa[2], fb[2];
#pragma unroll
        for (int mt = 0; mt < 2; ++mt) {
            const int rA = wm * 32 + mt * 16 + lm;
            fa[mt].q[0] = *(const u32x4*)&As[rA][hl * 8];
            fa[mt].q[1] = *(const u32x4*)&As[rA][hl * 8 + 16];
        }
#pragma unroll
        for (int nt = 0; nt < 2; ++nt) {
            const int cB = wn * 32 + nt * 16 + lm;
            fb[nt].q[0] = *(const u32x4*)&Bs[cB][hl * 8];
            fb[nt].q[1] = *(const u32x4*)&Bs[cB][hl * 8 + 16];
        }
#pragma unroll
        for (int mt = 0; mt < 2; ++mt)
#pragma unroll
            for (int nt = 0; nt < 2; ++nt)
                acc[mt][nt] = wmma_bf16(fa[mt].v, fb[nt].v, acc[mt][nt]);
        __syncthreads();
    }

    // epilogue: C layout -> row = v + 8*hl, col = lm (within 16x16 tile)
#pragma unroll
    for (int mt = 0; mt < 2; ++mt) {
#pragma unroll
        for (int nt = 0; nt < 2; ++nt) {
            const int c  = n0 + wn * 32 + nt * 16 + lm;
            const float bc = bias[c];
#pragma unroll
            for (int e = 0; e < 8; ++e) {
                const int r = m0 + wm * 32 + mt * 16 + e + 8 * hl;
                float val = acc[mt][nt][e] + bc;
                if constexpr (EPI == 0) {
                    outb[(size_t)r * N + c] = to_bf16(val);
                } else if constexpr (EPI == 1) {
                    outf[(size_t)r * N + c] = resid[(size_t)r * N + c] + val;
                } else if constexpr (EPI == 2) {
                    outf[(size_t)r * N + c] = val;
                } else {
                    const float ge = 0.5f * val *
                        (1.0f + erff(val * 0.70710678118654752f));
                    outb[(size_t)r * N + c] = to_bf16(ge);
                }
            }
        }
    }
}

// ---------------------------------------------------------------------------
// Fused causal attention, flash-style. One workgroup per (b,h); 8 waves;
// each wave owns 32 query rows. scale = 1/sqrt(E) (reference divides by
// sqrt(512), not sqrt(head_dim)). Causal masking only touches the diagonal
// key block (j == wave), so the mask compares live outside the hot path.
// ---------------------------------------------------------------------------
__global__ __launch_bounds__(256)
void k_attn(const __bf16* __restrict__ qg, const __bf16* __restrict__ kg,
            const __bf16* __restrict__ vg, __bf16* __restrict__ og) {
    __shared__ alignas(16) __bf16 Ks[T_SZ][72];    // [s][d]   36864 B
    __shared__ alignas(16) __bf16 Vt[D_SZ][264];   // [d][s]   33792 B
    __shared__ alignas(16) __bf16 Px[8][2][16 * 32]; // per-wave P bounce, 16 KB

    const int bh = blockIdx.x;
    const int b  = bh >> 3, h = bh & 7;
    const int tid  = threadIdx.x;
    const int wave = tid >> 5, lane = tid & 31;
    const int hl   = lane >> 4, lm = lane & 15;

    // ---- stage K (row-major) and V (transposed) into LDS ----
    {
        const int s = tid;
        union { u32x4 u[8]; __bf16 e[64]; } tk, tv;
        const __bf16* kr = kg + ((size_t)(b * T_SZ + s) * E_SZ + h * D_SZ);
        const __bf16* vr = vg + ((size_t)(b * T_SZ + s) * E_SZ + h * D_SZ);
#pragma unroll
        for (int i = 0; i < 8; ++i) {
            tk.u[i] = ((const u32x4*)kr)[i];
            tv.u[i] = ((const u32x4*)vr)[i];
        }
#pragma unroll
        for (int i = 0; i < 8; ++i) *(u32x4*)&Ks[s][i * 8] = tk.u[i];
#pragma unroll
        for (int d = 0; d < 64; ++d) Vt[d][s] = tv.e[d];
    }
    __syncthreads();

    // ---- Q fragments for this wave's 32 rows (2 m-tiles x 2 k-steps) ----
    FragAB fq[2][2];
#pragma unroll
    for (int mt = 0; mt < 2; ++mt) {
        const int r = wave * 32 + mt * 16 + lm;
        const __bf16* qr = qg + ((size_t)(b * T_SZ + r) * E_SZ + h * D_SZ);
#pragma unroll
        for (int ks = 0; ks < 2; ++ks) {
            const int k0 = ks * 32 + hl * 8;
            fq[mt][ks].q[0] = *(const u32x4*)(qr + k0);
            fq[mt][ks].q[1] = *(const u32x4*)(qr + k0 + 16);
        }
    }

    float mrow[2][8], lrow[2][8];
    v8f   O[2][4];
#pragma unroll
    for (int mt = 0; mt < 2; ++mt) {
#pragma unroll
        for (int e = 0; e < 8; ++e) { mrow[mt][e] = -3.0e38f; lrow[mt][e] = 0.f; }
#pragma unroll
        for (int dt = 0; dt < 4; ++dt) O[mt][dt] = vzero8();
    }

    const float scale = 0.044194173824159216f;   // 1/sqrt(512)

    for (int j = 0; j <= wave; ++j) {            // 32-key blocks (causal)
        const int sb = j * 32;

        // ---- S = Q x K^T  (2 m-tiles x 2 n-tiles, K=64 in two steps) ----
        FragAB fk[2][2];
#pragma unroll
        for (int nt = 0; nt < 2; ++nt) {
            const int s = sb + nt * 16 + lm;
#pragma unroll
            for (int ks = 0; ks < 2; ++ks) {
                const int k0 = ks * 32 + hl * 8;
                fk[nt][ks].q[0] = *(const u32x4*)&Ks[s][k0];
                fk[nt][ks].q[1] = *(const u32x4*)&Ks[s][k0 + 16];
            }
        }
        v8f S[2][2];
#pragma unroll
        for (int mt = 0; mt < 2; ++mt)
#pragma unroll
            for (int nt = 0; nt < 2; ++nt) {
                v8f acc = vzero8();
                acc = wmma_bf16(fq[mt][0].v, fk[nt][0].v, acc);
                acc = wmma_bf16(fq[mt][1].v, fk[nt][1].v, acc);
                S[mt][nt] = acc;
            }

        // ---- scale (+ causal mask only on the diagonal block) ----
        float P[2][2][8];
#pragma unroll
        for (int mt = 0; mt < 2; ++mt)
#pragma unroll
            for (int nt = 0; nt < 2; ++nt)
#pragma unroll
                for (int e = 0; e < 8; ++e)
                    P[mt][nt][e] = S[mt][nt][e] * scale;

        if (j == wave) {                         // wave-uniform branch
#pragma unroll
            for (int mt = 0; mt < 2; ++mt)
#pragma unroll
                for (int nt = 0; nt < 2; ++nt) {
                    const int sg = sb + nt * 16 + lm;
#pragma unroll
                    for (int e = 0; e < 8; ++e) {
                        const int r = wave * 32 + mt * 16 + e + 8 * hl;
                        if (sg > r) P[mt][nt][e] = -1.0e30f;
                    }
                }
        }

        // ---- online softmax (row reductions across 16-lane half-group) ----
#pragma unroll
        for (int mt = 0; mt < 2; ++mt) {
#pragma unroll
            for (int e = 0; e < 8; ++e) {
                float bm = fmaxf(P[mt][0][e], P[mt][1][e]);
                bm = fmaxf(bm, __shfl_xor(bm, 1, 32));
                bm = fmaxf(bm, __shfl_xor(bm, 2, 32));
                bm = fmaxf(bm, __shfl_xor(bm, 4, 32));
                bm = fmaxf(bm, __shfl_xor(bm, 8, 32));
                const float mnew  = fmaxf(mrow[mt][e], bm);
                const float alpha = __expf(mrow[mt][e] - mnew);
                mrow[mt][e] = mnew;
                const float p0 = __expf(P[mt][0][e] - mnew);
                const float p1 = __expf(P[mt][1][e] - mnew);
                P[mt][0][e] = p0; P[mt][1][e] = p1;
                float ps = p0 + p1;
                ps += __shfl_xor(ps, 1, 32);
                ps += __shfl_xor(ps, 2, 32);
                ps += __shfl_xor(ps, 4, 32);
                ps += __shfl_xor(ps, 8, 32);
                lrow[mt][e] = lrow[mt][e] * alpha + ps;
#pragma unroll
                for (int dt = 0; dt < 4; ++dt) O[mt][dt][e] *= alpha;
            }
        }

        // ---- re-layout P (C-layout f32 -> A-layout bf16) via LDS bounce ----
#pragma unroll
        for (int mt = 0; mt < 2; ++mt)
#pragma unroll
            for (int nt = 0; nt < 2; ++nt)
#pragma unroll
                for (int e = 0; e < 8; ++e) {
                    const int ml = e + 8 * hl;
                    Px[wave][mt][ml * 32 + nt * 16 + lm] = to_bf16(P[mt][nt][e]);
                }
        asm volatile("s_wait_dscnt 0x0" ::: "memory");  // same-wave LDS RAW

        FragAB fp[2];
#pragma unroll
        for (int mt = 0; mt < 2; ++mt) {
            fp[mt].q[0] = *(const u32x4*)&Px[wave][mt][lm * 32 + hl * 8];
            fp[mt].q[1] = *(const u32x4*)&Px[wave][mt][lm * 32 + hl * 8 + 16];
        }

        // ---- O += P x V  (K = 32 keys of this block) ----
#pragma unroll
        for (int dt = 0; dt < 4; ++dt) {
            FragAB fv;
            const int dc = dt * 16 + lm;
            fv.q[0] = *(const u32x4*)&Vt[dc][sb + hl * 8];
            fv.q[1] = *(const u32x4*)&Vt[dc][sb + hl * 8 + 16];
#pragma unroll
            for (int mt = 0; mt < 2; ++mt)
                O[mt][dt] = wmma_bf16(fp[mt].v, fv.v, O[mt][dt]);
        }
    }

    // ---- normalize and write bf16 output ----
#pragma unroll
    for (int mt = 0; mt < 2; ++mt) {
#pragma unroll
        for (int e = 0; e < 8; ++e) {
            const int r = wave * 32 + mt * 16 + e + 8 * hl;
            const float inv = 1.0f / lrow[mt][e];
            __bf16* orow = og + ((size_t)(b * T_SZ + r) * E_SZ + h * D_SZ);
#pragma unroll
            for (int dt = 0; dt < 4; ++dt)
                orow[dt * 16 + lm] = to_bf16(O[mt][dt][e] * inv);
        }
    }
}

// ---------------------------------------------------------------------------
// Host driver
// ---------------------------------------------------------------------------
extern "C" void kernel_launch(void* const* d_in, const int* in_sizes, int n_in,
                              void* d_out, int out_size, void* d_ws, size_t ws_size,
                              hipStream_t stream) {
    (void)in_sizes; (void)n_in; (void)out_size; (void)ws_size;

    const int*   idx   = (const int*)  d_in[0];
    const float* tok   = (const float*)d_in[1];
    const float* pos   = (const float*)d_in[2];
    const float* ln1g  = (const float*)d_in[3];
    const float* ln1b  = (const float*)d_in[4];
    const float* Wq    = (const float*)d_in[5];
    const float* bq    = (const float*)d_in[6];
    const float* Wk    = (const float*)d_in[7];
    const float* bk    = (const float*)d_in[8];
    const float* Wv    = (const float*)d_in[9];
    const float* bv    = (const float*)d_in[10];
    const float* Wo    = (const float*)d_in[11];
    const float* bo    = (const float*)d_in[12];
    const float* ln2g  = (const float*)d_in[13];
    const float* ln2b  = (const float*)d_in[14];
    const float* W1    = (const float*)d_in[15];
    const float* b1    = (const float*)d_in[16];
    const float* W2    = (const float*)d_in[17];
    const float* b2    = (const float*)d_in[18];
    const float* lnfg  = (const float*)d_in[19];
    const float* lnfb  = (const float*)d_in[20];
    const float* Wh    = (const float*)d_in[21];
    const float* bh    = (const float*)d_in[22];

    char*  ws  = (char*)d_ws;
    size_t off = 0;
    auto take = [&](size_t bytes) -> char* {
        char* p = ws + off;
        off = (off + bytes + 255) & ~(size_t)255;
        return p;
    };

    const size_t nEE = (size_t)L_SZ * E_SZ * E_SZ;     // 3.1M
    const size_t nEF = (size_t)L_SZ * E_SZ * FF_SZ;    // 12.6M
    const size_t nEV = (size_t)E_SZ * V_SZ;            // 4.2M

    __bf16* wqB = (__bf16*)take(nEE * 2);
    __bf16* wkB = (__bf16*)take(nEE * 2);
    __bf16* wvB = (__bf16*)take(nEE * 2);
    __bf16* woB = (__bf16*)take(nEE * 2);
    __bf16* w1B = (__bf16*)take(nEF * 2);
    __bf16* w2B = (__bf16*)take(nEF * 2);
    __bf16* whB = (__bf16*)take(nEV * 2);

    float*  xA  = (float*) take((size_t)M_TOK * E_SZ * 4);
    float*  xB  = (float*) take((size_t)M_TOK * E_SZ * 4);
    __bf16* hN  = (__bf16*)take((size_t)M_TOK * E_SZ * 2);
    __bf16* qB  = (__bf16*)take((size_t)M_TOK * E_SZ * 2);
    __bf16* kB  = (__bf16*)take((size_t)M_TOK * E_SZ * 2);
    __bf16* vB  = (__bf16*)take((size_t)M_TOK * E_SZ * 2);
    __bf16* aoB = (__bf16*)take((size_t)M_TOK * E_SZ * 2);
    __bf16* mhB = (__bf16*)take((size_t)M_TOK * FF_SZ * 2);

    auto cvt = [&](const float* s, __bf16* d, size_t n) {
        k_f32_to_bf16<<<(unsigned)((n + 255) / 256), 256, 0, stream>>>(s, d, (long long)n);
    };
    cvt(Wq, wqB, nEE); cvt(Wk, wkB, nEE); cvt(Wv, wvB, nEE); cvt(Wo, woB, nEE);
    cvt(W1, w1B, nEF); cvt(W2, w2B, nEF); cvt(Wh, whB, nEV);

    k_embed<<<(M_TOK * E_SZ) / 256, 256, 0, stream>>>(idx, tok, pos, xA);

    const dim3 blk(256);
    const dim3 gE(E_SZ / 64, M_TOK / 128);
    const dim3 gF(FF_SZ / 64, M_TOK / 128);
    const dim3 gV(V_SZ / 64, M_TOK / 128);

    for (int l = 0; l < L_SZ; ++l) {
        const size_t oEE = (size_t)l * E_SZ * E_SZ;
        const size_t oEF = (size_t)l * E_SZ * FF_SZ;

        k_ln_bf16<<<M_TOK / 8, blk, 0, stream>>>(xA, ln1g + l * E_SZ, ln1b + l * E_SZ, hN);
        k_gemm<0><<<gE, blk, 0, stream>>>(hN, wqB + oEE, bq + l * E_SZ, nullptr, qB, nullptr, M_TOK, E_SZ, E_SZ);
        k_gemm<0><<<gE, blk, 0, stream>>>(hN, wkB + oEE, bk + l * E_SZ, nullptr, kB, nullptr, M_TOK, E_SZ, E_SZ);
        k_gemm<0><<<gE, blk, 0, stream>>>(hN, wvB + oEE, bv + l * E_SZ, nullptr, vB, nullptr, M_TOK, E_SZ, E_SZ);
        k_attn<<<B_SZ * H_SZ, blk, 0, stream>>>(qB, kB, vB, aoB);
        k_gemm<1><<<gE, blk, 0, stream>>>(aoB, woB + oEE, bo + l * E_SZ, xA, nullptr, xB, M_TOK, E_SZ, E_SZ);
        k_ln_bf16<<<M_TOK / 8, blk, 0, stream>>>(xB, ln2g + l * E_SZ, ln2b + l * E_SZ, hN);
        k_gemm<3><<<gF, blk, 0, stream>>>(hN, w1B + oEF, b1 + l * FF_SZ, nullptr, mhB, nullptr, M_TOK, FF_SZ, E_SZ);
        k_gemm<1><<<gE, blk, 0, stream>>>(mhB, w2B + oEF, b2 + l * E_SZ, xB, nullptr, xA, M_TOK, E_SZ, FF_SZ);
    }

    k_ln_bf16<<<M_TOK / 8, blk, 0, stream>>>(xA, lnfg, lnfb, hN);
    k_gemm<2><<<gV, blk, 0, stream>>>(hN, whB, bh, nullptr, nullptr, (float*)d_out,
                                      M_TOK, V_SZ, E_SZ);
}
